// ObjectImplicitNetworkGrid_21766894256738
// MI455X (gfx1250) — matrix operand order repaired and checked by reading, not compiled, not run on hardware
//
#include <hip/hip_runtime.h>
#include <cstdint>
#include <cstddef>

// ---------------------------------------------------------------------------
// Fully-fused instant-NGP hashgrid encode + 5-layer MLP for gfx1250 (MI455X).
// f16 WMMA (v_wmma_f32_16x16x32_f16), activations resident in LDS,
// weights converted once to f16 in d_ws (L2-resident, ~0.6 MB).
// ---------------------------------------------------------------------------

typedef _Float16 v8h  __attribute__((ext_vector_type(8)));
typedef _Float16 v16h __attribute__((ext_vector_type(16)));
typedef float    v8f  __attribute__((ext_vector_type(8)));

#define LVLS 16
#define TBL  (1u << 19)            // hashmap entries per level
#define ACT_STRIDE 264             // 256 + 8 halves pad (bank-conflict dodge)
#define WPG 4                      // waves per workgroup
#define ROWS_PER_WG (WPG * 16)

// Per-level grid resolutions: ceil(16 * (2^(7/15))^l), l = 0..15
__constant__ int c_res[LVLS] = {16, 23, 31, 43, 59, 81, 112, 155,
                                213, 295, 407, 562, 777, 1073, 1483, 2048};

// f16 weight workspace layout (element offsets)
#define W0H_OFF 0                        // [256][64]  (K padded 35->64, feat shift)
#define W1H_OFF (W0H_OFF + 256 * 64)     // [256][256]
#define W2H_OFF (W1H_OFF + 256 * 256)
#define W3H_OFF (W2H_OFF + 256 * 256)
#define W4H_OFF (W3H_OFF + 256 * 256)    // [272][256] (N padded 264->272)
#define WS_TOTAL (W4H_OFF + 272 * 256)   // 282,624 halves = 565,248 bytes

// ---------------------------------------------------------------------------
// Prep: convert f32 weights -> padded f16 tables in workspace
// ---------------------------------------------------------------------------
__global__ void __launch_bounds__(256) ngp_prep_weights(
    const float* __restrict__ W0, const float* __restrict__ W1,
    const float* __restrict__ W2, const float* __restrict__ W3,
    const float* __restrict__ W4, _Float16* __restrict__ ws) {
  int i = blockIdx.x * 256 + threadIdx.x;
  if (i >= WS_TOTAL) return;
  if (i < W1H_OFF) {                       // W0h [256][64]
    int n = i >> 6, k = i & 63;
    float v = 0.f;
    if (k < 3) v = W0[n * 35 + k];                 // raw xyz
    else if (k >= 4 && k < 36) v = W0[n * 35 + (k - 1)];  // hash feats shifted +1
    ws[i] = (_Float16)v;
  } else if (i < W2H_OFF) {
    int j = i - W1H_OFF; ws[i] = (_Float16)W1[j];
  } else if (i < W3H_OFF) {
    int j = i - W2H_OFF; ws[i] = (_Float16)W2[j];
  } else if (i < W4H_OFF) {
    int j = i - W3H_OFF; ws[i] = (_Float16)W3[j];
  } else {                                  // W4h [272][256], rows >= 264 zero
    int j = i - W4H_OFF;
    int n = j >> 8, k = j & 255;
    ws[i] = (_Float16)((n < 264) ? W4[n * 256 + k] : 0.f);
  }
}

// ---------------------------------------------------------------------------
// Fragment loaders (WMMA 16x16x32 f16 VGPR layouts, wave32)
// ---------------------------------------------------------------------------
// A (16x32, M x K): lane<16 -> row=lane, v0..3: K=ks*32+0..7, v4..7: K=ks*32+16..23
//                   lane>=16 -> same row set, +8 on K within each half.
__device__ __forceinline__ v16h load_a_frag(const _Float16* actRow, int ks, int hi) {
  const v8h lo = *(const v8h*)(actRow + ks * 32 + 8 * hi);
  const v8h hv = *(const v8h*)(actRow + ks * 32 + 16 + 8 * hi);
  return __builtin_shufflevector(lo, hv, 0, 1, 2, 3, 4, 5, 6, 7,
                                 8, 9, 10, 11, 12, 13, 14, 15);
}

// B (32x16, K x N): lane holds col n; K = ks*32+8*hi+0..7 and ks*32+16+8*hi+0..7
// (same K order as A) from row-major W[n][k] (computes h @ W^T).
__device__ __forceinline__ v16h load_b_frag(const _Float16* W, int Kd, int n,
                                            int ks, int hi) {
  const _Float16* p = W + (size_t)n * Kd + ks * 32 + 8 * hi;
  const v8h lo = *(const v8h*)(p);
  const v8h hv = *(const v8h*)(p + 16);
  return __builtin_shufflevector(lo, hv, 0, 1, 2, 3, 4, 5, 6, 7,
                                 8, 9, 10, 11, 12, 13, 14, 15);
}

// softplus(100*t)/100, branchless & overflow-free:
//   softplus(u) = max(u,0) + log1p(exp(-|u|))
// exp arg <= 0 (no overflow), log arg in (1,2] (no denorm/inf) -> safe to use
// raw v_exp_f32 / v_log_f32 (TRANS ops co-execute with the WMMA pipe).
__device__ __forceinline__ float softplus100(float t) {
  float u = 100.f * t;
  float e = __builtin_amdgcn_exp2f(fabsf(u) * -1.44269504089f);  // exp(-|u|)
  float l = __builtin_amdgcn_logf(1.f + e) * 0.69314718056f;     // ln(1+e)
  return fmaxf(t, 0.f) + l * 0.01f;
}

__device__ __forceinline__ uint32_t pack_h2(float a, float b) {
  union { _Float16 h[2]; uint32_t u; } t;
  t.h[0] = (_Float16)a; t.h[1] = (_Float16)b;
  return t.u;
}

// ---------------------------------------------------------------------------
// One hidden layer: act(16 x K) @ W^T (K x 256) + b -> softplus -> act (f16, LDS)
// A fragments cached in registers before outputs overwrite the buffer.
// ---------------------------------------------------------------------------
template <int KSTEPS>
__device__ __forceinline__ void mlp_layer(_Float16* act, const _Float16* W,
                                          const float* bias, int lane) {
  const int n16 = lane & 15, hi = lane >> 4;
  const _Float16* arow = act + n16 * ACT_STRIDE;   // A row = lane&15
  v16h A[KSTEPS];
#pragma unroll
  for (int s = 0; s < KSTEPS; ++s) A[s] = load_a_frag(arow, s, hi);

#pragma unroll 1
  for (int j = 0; j < 16; ++j) {                   // 16 output N-tiles
    const int n = j * 16 + n16;
    v8f acc = {};
#pragma unroll
    for (int s = 0; s < KSTEPS; ++s) {
      v16h B = load_b_frag(W, KSTEPS * 32, n, s, hi);
      acc = __builtin_amdgcn_wmma_f32_16x16x32_f16(false, A[s], false, B,
                                                   (short)0, acc, false, false);
    }
    const float bv = bias[n];
#pragma unroll
    for (int r = 0; r < 8; ++r) {                  // D: vgpr r -> row m=r+8*hi
      float sp = softplus100(acc[r] + bv);
      act[(r + 8 * hi) * ACT_STRIDE + n] = (_Float16)sp;
    }
  }
}

// Output layer: 256 -> 264 (N padded to 272 = 17 tiles), f32 store to HBM.
__device__ __forceinline__ void mlp_out(const _Float16* act, const _Float16* W4,
                                        const float* b4, float* outBase, int lane) {
  const int n16 = lane & 15, hi = lane >> 4;
  const _Float16* arow = act + n16 * ACT_STRIDE;
  v16h A[8];
#pragma unroll
  for (int s = 0; s < 8; ++s) A[s] = load_a_frag(arow, s, hi);

#pragma unroll 1
  for (int j = 0; j < 17; ++j) {
    const int n = j * 16 + n16;
    v8f acc = {};
#pragma unroll
    for (int s = 0; s < 8; ++s) {
      v16h B = load_b_frag(W4, 256, n, s, hi);
      acc = __builtin_amdgcn_wmma_f32_16x16x32_f16(false, A[s], false, B,
                                                   (short)0, acc, false, false);
    }
    if (n < 264) {
      const float bv = b4[n];
#pragma unroll
      for (int r = 0; r < 8; ++r)
        outBase[(size_t)(r + 8 * hi) * 264 + n] = acc[r] + bv;
    }
  }
}

// ---------------------------------------------------------------------------
// Fused kernel: hash-grid encode into LDS (f16), then 5 WMMA layers.
// Block = 128 threads = 4 waves; each wave owns 16 rows.
// ---------------------------------------------------------------------------
__global__ void __launch_bounds__(WPG * 32) ngp_fused(
    const float* __restrict__ X, const float* __restrict__ tables,
    const _Float16* __restrict__ ws,
    const float* __restrict__ b0, const float* __restrict__ b1,
    const float* __restrict__ b2, const float* __restrict__ b3,
    const float* __restrict__ b4, float* __restrict__ out) {
  const int lane = threadIdx.x & 31;
  const int wave = threadIdx.x >> 5;
  const int rowBase = (blockIdx.x * WPG + wave) * 16;

  __shared__ _Float16 s_act[WPG][16][ACT_STRIDE];
  _Float16* act = &s_act[wave][0][0];

  // ---- phase 0: raw x into cols 0..2, zero pad cols 3 and 36..63 ----------
  if (lane < 16) {
    const int r = lane;
    const float* xr = X + (size_t)(rowBase + r) * 3;
    uint32_t* p = (uint32_t*)&s_act[wave][r][0];
    p[0] = pack_h2(xr[0], xr[1]);
    p[1] = pack_h2(xr[2], 0.f);
#pragma unroll
    for (int c = 36; c < 64; c += 2) p[c >> 1] = 0u;
  }

  // ---- phase 1: hash-grid encode; lane = (level, row-parity), 8 iters -----
#pragma unroll 1
  for (int it = 0; it < 8; ++it) {
    const int row = it * 2 + (lane >> 4);
    const int level = lane & 15;
    const float* xr = X + (size_t)(rowBase + row) * 3;
    const int res = c_res[level];
    const float rm1 = (float)(res - 1);
    // x01 = clamp((x/1.5 + 1)*0.5, 0, 1) = clamp(x/3 + 0.5, 0, 1)
    float p0 = fminf(fmaxf(xr[0] * (1.f / 3.f) + 0.5f, 0.f), 1.f) * rm1;
    float p1 = fminf(fmaxf(xr[1] * (1.f / 3.f) + 0.5f, 0.f), 1.f) * rm1;
    float p2 = fminf(fmaxf(xr[2] * (1.f / 3.f) + 0.5f, 0.f), 1.f) * rm1;
    float fl0 = floorf(p0), fl1 = floorf(p1), fl2 = floorf(p2);
    float f0 = p0 - fl0, f1 = p1 - fl1, f2 = p2 - fl2;
    uint32_t i0 = (uint32_t)fl0, i1 = (uint32_t)fl1, i2 = (uint32_t)fl2;
    const bool dense = ((long long)res * res * res) <= (long long)TBL;
    float a0 = 0.f, a1 = 0.f;
#pragma unroll
    for (int c = 0; c < 8; ++c) {
      const uint32_t ox = (c >> 2) & 1, oy = (c >> 1) & 1, oz = c & 1;
      const uint32_t cx = i0 + ox, cy = i1 + oy, cz = i2 + oz;
      const float w = (ox ? f0 : 1.f - f0) * (oy ? f1 : 1.f - f1) *
                      (oz ? f2 : 1.f - f2);
      uint32_t idx = dense
          ? (cx + cy * (uint32_t)res + cz * (uint32_t)(res * res))
          : (cx ^ (cy * 2654435761u) ^ (cz * 805459861u));
      idx &= (TBL - 1u);
      const float2 f = *(const float2*)(tables + ((size_t)level * TBL + idx) * 2);
      a0 += w * f.x;
      a1 += w * f.y;
    }
    // features at even columns 4 + 2*level (aligned half2 store)
    *(uint32_t*)&s_act[wave][row][4 + 2 * level] = pack_h2(a0, a1);
  }
  __syncthreads();

  // ---- phase 2: MLP through WMMA --------------------------------------
  const _Float16* W0h = ws + W0H_OFF;
  const _Float16* W1h = ws + W1H_OFF;
  const _Float16* W2h = ws + W2H_OFF;
  const _Float16* W3h = ws + W3H_OFF;
  const _Float16* W4h = ws + W4H_OFF;

  mlp_layer<2>(act, W0h, b0, lane);   // 64 -> 256
  __syncthreads();
  mlp_layer<8>(act, W1h, b1, lane);   // 256 -> 256
  __syncthreads();
  mlp_layer<8>(act, W2h, b2, lane);
  __syncthreads();
  mlp_layer<8>(act, W3h, b3, lane);
  __syncthreads();
  mlp_out(act, W4h, b4, out + (size_t)rowBase * 264, lane);  // 256 -> 264
}

// ---------------------------------------------------------------------------
extern "C" void kernel_launch(void* const* d_in, const int* in_sizes, int n_in,
                              void* d_out, int out_size, void* d_ws, size_t ws_size,
                              hipStream_t stream) {
  const float* x      = (const float*)d_in[0];
  const float* tables = (const float*)d_in[1];
  const float* W0 = (const float*)d_in[2];  const float* b0 = (const float*)d_in[3];
  const float* W1 = (const float*)d_in[4];  const float* b1 = (const float*)d_in[5];
  const float* W2 = (const float*)d_in[6];  const float* b2 = (const float*)d_in[7];
  const float* W3 = (const float*)d_in[8];  const float* b3 = (const float*)d_in[9];
  const float* W4 = (const float*)d_in[10]; const float* b4 = (const float*)d_in[11];
  _Float16* ws = (_Float16*)d_ws;
  float* out = (float*)d_out;

  const int n = in_sizes[0] / 3;             // 131072 points

  // Convert/pad weights to f16 (deterministic every call; graph-safe).
  const int prepBlocks = (WS_TOTAL + 255) / 256;
  ngp_prep_weights<<<prepBlocks, 256, 0, stream>>>(W0, W1, W2, W3, W4, ws);

  // Fused encode + MLP: 64 rows per 128-thread block.
  const int blocks = (n + ROWS_PER_WG - 1) / ROWS_PER_WG;
  ngp_fused<<<blocks, WPG * 32, 0, stream>>>(x, tables, ws, b0, b1, b2, b3, b4, out);
}
